// MultiHeadAttention_57758720196645
// MI455X (gfx1250) — matrix operand order repaired and checked by reading, not compiled, not run on hardware
//
#include <hip/hip_runtime.h>

#define H_   768
#define NH_  12
#define HD_  64
#define T_   512
#define B_   16
#define MT_  (B_ * T_)     // 8192 token rows
#define TQ_  32            // query tile per attention block
#define LORA_SCALE 2.0f    // alpha/r = 8/4

typedef __attribute__((ext_vector_type(16))) __bf16 v16bf;
typedef __attribute__((ext_vector_type(8)))  __bf16 v8bf;
typedef __attribute__((ext_vector_type(8)))  float  v8f;

union BF16x16 { v16bf v; v8bf h[2]; };

// A-fragment (16x32 bf16, ISA 7.12.2): lane holds two 16B K-chunks at +0 and +16
__device__ __forceinline__ v16bf ld_afrag(const __bf16* p) {
  BF16x16 u;
  u.h[0] = *(const v8bf*)(p);
  u.h[1] = *(const v8bf*)(p + 16);
  return u.v;
}

__device__ __forceinline__ v8f wmma_bf16f32(v16bf a, v16bf b, v8f c) {
  return __builtin_amdgcn_wmma_f32_16x16x32_bf16(false, a, false, b, (short)0, c,
                                                 false, false);
}

// ---------------------------------------------------------------------------
// prep: bf16 convert, optionally adding rank-4 LoRA update W + scale*(Bl @ Al)
// Al: [4, cols], Bl: [rows, 4]
// ---------------------------------------------------------------------------
__global__ void __launch_bounds__(256) prep_bf16_kernel(
    const float* __restrict__ W, const float* __restrict__ Al,
    const float* __restrict__ Bl, __bf16* __restrict__ outp,
    int rows, int cols, float scale)
{
  int idx = blockIdx.x * 256 + threadIdx.x;
  if (idx >= rows * cols) return;
  int r = idx / cols, c = idx % cols;
  float v = W[idx];
  if (Al != nullptr) {
    float acc = 0.0f;
#pragma unroll
    for (int j = 0; j < 4; ++j) acc += Bl[r * 4 + j] * Al[j * cols + c];
    v += scale * acc;
  }
  outp[idx] = (__bf16)v;
}

// ---------------------------------------------------------------------------
// GEMM: out[m,n] = sum_k A[m,k]*Bw[n,k] + bias[n]      (A: [M,K], Bw: [N,K])
// block = 256 thr (8 wave32), block tile 128x128, wave tile 32x64 (2x4 WMMA)
// MODE 0: f32 row-major out   MODE 1: bf16 row-major out
// MODE 2: bf16 out scattered into V-transposed layout [b, h, d, t]
// ---------------------------------------------------------------------------
template <int MODE>
__global__ void __launch_bounds__(256) gemm_bf16_kernel(
    const __bf16* __restrict__ A, const __bf16* __restrict__ Bw,
    const float* __restrict__ bias, void* __restrict__ outp,
    int M, int N, int K)
{
  const int lane = threadIdx.x & 31;
  const int w    = threadIdx.x >> 5;
  const int l16  = lane & 15;
  const int hl   = lane >> 4;
  const int wm   = w >> 1;                     // 0..3
  const int wn   = w & 1;                      // 0..1
  const int m0   = blockIdx.x * 128 + wm * 32;
  const int n0   = blockIdx.y * 128 + wn * 64;

  v8f acc[2][4] = {};

  const __bf16* arow[2];
  const __bf16* brow[4];
#pragma unroll
  for (int t = 0; t < 2; ++t)
    arow[t] = A + (size_t)(m0 + t * 16 + l16) * K;
#pragma unroll
  for (int t = 0; t < 4; ++t)
    brow[t] = Bw + (size_t)(n0 + t * 16 + l16) * K;

  for (int kk = 0; kk < K; kk += 32) {
    v16bf af[2], bf2[4];
#pragma unroll
    for (int t = 0; t < 2; ++t)
      af[t] = ld_afrag(arow[t] + kk + hl * 8);
#pragma unroll
    for (int t = 0; t < 4; ++t)
      bf2[t] = ld_afrag(brow[t] + kk + hl * 8);  // same K order as A-fragment
    if (kk + 32 < K) {  // CDNA5 global_prefetch_b8 for next K-slab
      __builtin_prefetch(arow[0] + kk + 32 + hl * 8, 0, 1);
      __builtin_prefetch(brow[0] + kk + 32 + hl * 16, 0, 1);
    }
#pragma unroll
    for (int ti = 0; ti < 2; ++ti)
#pragma unroll
      for (int tj = 0; tj < 4; ++tj)
        acc[ti][tj] = wmma_bf16f32(af[ti], bf2[tj], acc[ti][tj]);
  }

#pragma unroll
  for (int ti = 0; ti < 2; ++ti)
#pragma unroll
    for (int tj = 0; tj < 4; ++tj)
#pragma unroll
      for (int r = 0; r < 8; ++r) {
        int m = m0 + ti * 16 + r + 8 * hl;     // C/D layout: row = r + 8*(lane>>4)
        int n = n0 + tj * 16 + l16;            //             col = lane&15
        float val = acc[ti][tj][r] + bias[n];
        if (MODE == 0) {
          ((float*)outp)[(size_t)m * N + n] = val;
        } else if (MODE == 1) {
          ((__bf16*)outp)[(size_t)m * N + n] = (__bf16)val;
        } else {  // V-transpose scatter: m = b*T+t, n = h*64+d -> [b,h,d,t]
          int bb = m >> 9, t = m & (T_ - 1);
          int hh = n >> 6, d = n & (HD_ - 1);
          ((__bf16*)outp)[(((size_t)bb * NH_ + hh) * HD_ + d) * T_ + t] = (__bf16)val;
        }
      }
}

// ---------------------------------------------------------------------------
// Attention: per (b, h, 32-query tile).  8 wave32 / block.
// Phase 1: S = (Q Kt)/8 + mask   (scores live in accumulator VGPRs)
// Phase 2: distributed softmax (shfl_xor within 16-lane row groups + LDS),
//          unnormalized probs -> padded LDS bf16 tile
// Phase 3: O = P @ V via WMMA (A from LDS, B from pre-transposed V), then
//          normalize by row-sum in the epilogue.
// ---------------------------------------------------------------------------
__global__ void __launch_bounds__(256) attn_kernel(
    const __bf16* __restrict__ qb, const __bf16* __restrict__ kb,
    const __bf16* __restrict__ vT, const int* __restrict__ mask,
    __bf16* __restrict__ attnb)
{
  const int qt = blockIdx.x;   // 0..15
  const int h  = blockIdx.y;   // 0..11
  const int b  = blockIdx.z;   // 0..15
  const int lane = threadIdx.x & 31;
  const int w    = threadIdx.x >> 5;
  const int l16  = lane & 15;
  const int hl   = lane >> 4;

  __shared__ __align__(16) __bf16 P[TQ_][520];  // pad 512->520 to stagger banks
  __shared__ float red[TQ_][4];
  __shared__ float rowsum[TQ_];

  const int wmt = w >> 2;   // 0..1 : 16-row m-tile
  const int wnq = w & 3;    // 0..3 : 128-key column quarter

  // ---- Phase 1: scores --------------------------------------------------
  v16bf aq[2];
  {
    int grow = b * T_ + qt * TQ_ + wmt * 16 + l16;
    const __bf16* qrow = qb + (size_t)grow * H_ + h * HD_;
#pragma unroll
    for (int ks = 0; ks < 2; ++ks)            // K = HD = 64 -> 2 ksteps
      aq[ks] = ld_afrag(qrow + ks * 32 + hl * 8);
  }
  v8f s[8] = {};
#pragma unroll
  for (int j = 0; j < 8; ++j) {
    int key = wnq * 128 + j * 16 + l16;
    const __bf16* krow = kb + (size_t)(b * T_ + key) * H_ + h * HD_;
#pragma unroll
    for (int ks = 0; ks < 2; ++ks) {
      v16bf bk = ld_afrag(krow + ks * 32 + hl * 8);
      s[j] = wmma_bf16f32(aq[ks], bk, s[j]);
    }
  }
#pragma unroll
  for (int j = 0; j < 8; ++j) {               // scale 1/sqrt(64) + key mask
    int key = wnq * 128 + j * 16 + l16;
    float mb = (mask[b * T_ + key] == 0) ? -1e9f : 0.0f;
#pragma unroll
    for (int r = 0; r < 8; ++r)
      s[j][r] = s[j][r] * 0.125f + mb;
  }

  // ---- Phase 2: softmax stats -------------------------------------------
  float rmax[8];
#pragma unroll
  for (int r = 0; r < 8; ++r) {
    float pm = -3.0e38f;
#pragma unroll
    for (int j = 0; j < 8; ++j) pm = fmaxf(pm, s[j][r]);
    pm = fmaxf(pm, __shfl_xor(pm, 1));
    pm = fmaxf(pm, __shfl_xor(pm, 2));
    pm = fmaxf(pm, __shfl_xor(pm, 4));
    pm = fmaxf(pm, __shfl_xor(pm, 8));        // 16-lane row group reduction
    if (l16 == 0) red[wmt * 16 + r + 8 * hl][wnq] = pm;
  }
  __syncthreads();
#pragma unroll
  for (int r = 0; r < 8; ++r) {
    int row = wmt * 16 + r + 8 * hl;
    rmax[r] = fmaxf(fmaxf(red[row][0], red[row][1]),
                    fmaxf(red[row][2], red[row][3]));
  }
  __syncthreads();
#pragma unroll
  for (int r = 0; r < 8; ++r) {
    int row = wmt * 16 + r + 8 * hl;
    float ps = 0.0f;
#pragma unroll
    for (int j = 0; j < 8; ++j) {
      float p = __expf(s[j][r] - rmax[r]);    // unnormalized prob
      ps += p;
      P[row][wnq * 128 + j * 16 + l16] = (__bf16)p;
    }
    ps += __shfl_xor(ps, 1);
    ps += __shfl_xor(ps, 2);
    ps += __shfl_xor(ps, 4);
    ps += __shfl_xor(ps, 8);
    if (l16 == 0) red[row][wnq] = ps;
  }
  __syncthreads();
  if (threadIdx.x < TQ_)
    rowsum[threadIdx.x] = red[threadIdx.x][0] + red[threadIdx.x][1] +
                          red[threadIdx.x][2] + red[threadIdx.x][3];
  __syncthreads();

  // ---- Phase 3: O = P @ V  (wave -> one 16x16 output tile) --------------
  const int mt = w >> 2;   // 0..1
  const int nt = w & 3;    // 0..3  (16 of 64 head dims)
  v8f o = {};
  const __bf16* vrow = vT + ((size_t)(b * NH_ + h) * HD_ + nt * 16 + l16) * T_;
#pragma unroll
  for (int ks = 0; ks < 16; ++ks) {           // K = T = 512 -> 16 steps of 32
    v16bf ap = ld_afrag(&P[mt * 16 + l16][ks * 32 + hl * 8]);       // ds_load_b128
    v16bf bv = ld_afrag(vrow + ks * 32 + hl * 8);
    o = wmma_bf16f32(ap, bv, o);
  }
  {
    int ncol = h * HD_ + nt * 16 + l16;
#pragma unroll
    for (int r = 0; r < 8; ++r) {
      int m = mt * 16 + r + 8 * hl;
      float val = o[r] / rowsum[m];           // deferred softmax normalization
      int grow = b * T_ + qt * TQ_ + m;
      attnb[(size_t)grow * H_ + ncol] = (__bf16)val;
    }
  }
}

// ---------------------------------------------------------------------------
extern "C" void kernel_launch(void* const* d_in, const int* in_sizes, int n_in,
                              void* d_out, int out_size, void* d_ws, size_t ws_size,
                              hipStream_t stream)
{
  const float* x  = (const float*)d_in[0];
  const int*   mk = (const int*)d_in[1];
  const float* Wq = (const float*)d_in[2];
  const float* bq = (const float*)d_in[3];
  const float* Aq = (const float*)d_in[4];
  const float* Bq = (const float*)d_in[5];
  const float* Wk = (const float*)d_in[6];
  const float* bk = (const float*)d_in[7];
  const float* Wv = (const float*)d_in[8];
  const float* bv = (const float*)d_in[9];
  const float* Av = (const float*)d_in[10];
  const float* Bv = (const float*)d_in[11];
  const float* Wo = (const float*)d_in[12];
  const float* bo = (const float*)d_in[13];
  float* out = (float*)d_out;
  (void)in_sizes; (void)n_in; (void)out_size; (void)ws_size;

  // workspace carve-out (~68 MB total)
  char* base = (char*)d_ws;
  size_t off = 0;
  auto alloc = [&](size_t elems) -> __bf16* {
    __bf16* p = (__bf16*)(base + off);
    off = (off + elems * sizeof(__bf16) + 255) & ~(size_t)255;
    return p;
  };
  __bf16* xb    = alloc((size_t)MT_ * H_);  // x in bf16
  __bf16* Wqe   = alloc((size_t)H_ * H_);   // Wq + s*Bq@Aq
  __bf16* Wkb   = alloc((size_t)H_ * H_);
  __bf16* Wve   = alloc((size_t)H_ * H_);   // Wv + s*Bv@Av
  __bf16* Wob   = alloc((size_t)H_ * H_);
  __bf16* qb    = alloc((size_t)MT_ * H_);
  __bf16* kbuf  = alloc((size_t)MT_ * H_);
  __bf16* vTb   = alloc((size_t)MT_ * H_);  // [b, h, d, t]
  __bf16* attnb = alloc((size_t)MT_ * H_);

  // precision prep + LoRA folding
  {
    int nx = MT_ * H_;
    prep_bf16_kernel<<<(nx + 255) / 256, 256, 0, stream>>>(
        x, nullptr, nullptr, xb, MT_, H_, 0.0f);
    int nw = H_ * H_;
    prep_bf16_kernel<<<(nw + 255) / 256, 256, 0, stream>>>(
        Wq, Aq, Bq, Wqe, H_, H_, LORA_SCALE);
    prep_bf16_kernel<<<(nw + 255) / 256, 256, 0, stream>>>(
        Wk, nullptr, nullptr, Wkb, H_, H_, 0.0f);
    prep_bf16_kernel<<<(nw + 255) / 256, 256, 0, stream>>>(
        Wv, Av, Bv, Wve, H_, H_, LORA_SCALE);
    prep_bf16_kernel<<<(nw + 255) / 256, 256, 0, stream>>>(
        Wo, nullptr, nullptr, Wob, H_, H_, 0.0f);
  }

  // projections (q/k row-major bf16; v written pre-transposed)
  dim3 gg(MT_ / 128, H_ / 128);
  gemm_bf16_kernel<1><<<gg, 256, 0, stream>>>(xb, Wqe, bq, qb,   MT_, H_, H_);
  gemm_bf16_kernel<1><<<gg, 256, 0, stream>>>(xb, Wkb, bk, kbuf, MT_, H_, H_);
  gemm_bf16_kernel<2><<<gg, 256, 0, stream>>>(xb, Wve, bv, vTb,  MT_, H_, H_);

  // attention
  attn_kernel<<<dim3(T_ / TQ_, NH_, B_), 256, 0, stream>>>(qb, kbuf, vTb, mk, attnb);

  // output projection -> f32
  gemm_bf16_kernel<0><<<gg, 256, 0, stream>>>(attnb, Wob, bo, out, MT_, H_, H_);
}
